// PMGAE_70239895159230
// MI455X (gfx1250) — matrix operand order repaired and checked by reading, not compiled, not run on hardware
//
#include <hip/hip_runtime.h>
#include <hip/hip_bf16.h>

#define IN_DIM  512
#define HID_DIM 256
#define OUT_DIM 128
#define N_IDX   8192

typedef __attribute__((ext_vector_type(16))) __bf16 v16bf;
typedef __attribute__((ext_vector_type(8)))  float  v8f;

static __device__ __forceinline__ unsigned short f2bf_bits(float x) {
    union { __bf16 b; unsigned short s; } c; c.b = (__bf16)x; return c.s;
}
static __device__ __forceinline__ __bf16 bits2bf(unsigned short s) {
    union { unsigned short s; __bf16 b; } c; c.s = s; return c.b;
}
static __device__ __forceinline__ unsigned int pack2bf(float x, float y) {
    union { __bf16 b[2]; unsigned int u; } c;
    c.b[0] = (__bf16)x; c.b[1] = (__bf16)y; return c.u;
}

// ---------------------------------------------------------------- degrees
__global__ void __launch_bounds__(256) degree_kernel(
    const int* __restrict__ src, const int* __restrict__ dst,
    float* __restrict__ dout, float* __restrict__ din, int n_edges)
{
    const int e = blockIdx.x * blockDim.x + threadIdx.x;
    if (e < n_edges) {
        atomicAdd(&dout[src[e]], 1.0f);
        atomicAdd(&din[dst[e]], 1.0f);
    }
}

__global__ void __launch_bounds__(256) rsqrt_norm_kernel(
    float* __restrict__ dout, float* __restrict__ din, int n)
{
    const int i = blockIdx.x * blockDim.x + threadIdx.x;
    if (i < n) {
        dout[i] = rsqrtf(fmaxf(dout[i], 1.0f));   // ns (source norm)
        din[i]  = rsqrtf(fmaxf(din[i],  1.0f));   // nd (dest norm)
    }
}

// ------------------------------------------------- T = (X * ns[:,None]) @ W
// Block: 256 threads = 8 waves; covers 16 rows x 128 cols of T.
// Per k-step: stage A tile (16x32, ns-scaled bf16) and W tile (32x128 bf16,
// stored K-major per column) in LDS with coalesced loads; each wave builds
// one 16x16 tile's fragments from LDS and issues v_wmma_f32_16x16x32_bf16.
// A frag: lane L holds row M=L&15; v[i] pair K = 2i + (i>=4?8:0) + (L>=16?8:0)
// B frag: lane L holds col N=L&15; v[i] pair K = 2i + (i>=4?8:0) + (L>=16?8:0)
template<int FIN, int FOUT>
__global__ void __launch_bounds__(256) gcn_gemm_wmma(
    const float* __restrict__ X, const float* __restrict__ ns,
    const float* __restrict__ W, float* __restrict__ T)
{
    __shared__ unsigned short sA[16 * 32];    // [row][k]
    __shared__ unsigned short sB[128 * 32];   // [col][k]  (transposed W tile)

    const int t    = threadIdx.x;
    const int wave = t >> 5;
    const int lane = t & 31;
    const int half = lane >> 4;
    const int m0   = blockIdx.x * 16;
    const int cb0  = blockIdx.y * 128;
    const int col  = cb0 + wave * 16 + (lane & 15);

    // per-thread A staging coords: 512 elems, 2 per thread (one float2)
    const int aIdx = t * 2;                   // linear elem index
    const int aRow = aIdx >> 5;               // 0..15
    const int aK   = aIdx & 31;               // even
    const float sRow = ns[m0 + aRow];
    const float* xrow = X + (size_t)(m0 + aRow) * FIN;

    v8f acc = {};
    for (int k0 = 0; k0 < FIN; k0 += 32) {
        // ---- stage A (coalesced float2 per thread, fold ns scale) ----
        const float2 xp = *(const float2*)(xrow + k0 + aK);
        *(unsigned int*)&sA[aRow * 32 + aK] = pack2bf(xp.x * sRow, xp.y * sRow);
        // ---- stage W tile: 32x128 elems, 16 per thread, coalesced ----
        #pragma unroll
        for (int j = 0; j < 16; ++j) {
            const int lin = t + j * 256;      // 0..4095
            const int r   = lin >> 7;         // k within tile
            const int c   = lin & 127;        // col within tile
            sB[c * 32 + r] = f2bf_bits(W[(size_t)(k0 + r) * FOUT + cb0 + c]);
        }
        __syncthreads();

        // ---- build fragments from LDS ----
        v16bf a, b;
        const int mloc = lane & 15;
        const int cloc = wave * 16 + (lane & 15);
        #pragma unroll
        for (int i = 0; i < 8; ++i) {
            const int ka = 2 * i + (i >= 4 ? 8 : 0) + (half ? 8 : 0);
            unsigned int ua = *(const unsigned int*)&sA[mloc * 32 + ka];
            a[2*i] = bits2bf((unsigned short)ua); a[2*i+1] = bits2bf((unsigned short)(ua >> 16));
            const int kb = 2 * i + (i >= 4 ? 8 : 0) + (half ? 8 : 0);
            unsigned int ub = *(const unsigned int*)&sB[cloc * 32 + kb];
            b[2*i] = bits2bf((unsigned short)ub); b[2*i+1] = bits2bf((unsigned short)(ub >> 16));
        }
        __syncthreads();   // LDS consumed; safe to restage next iter

        acc = __builtin_amdgcn_wmma_f32_16x16x32_bf16(
            false, a, false, b, (short)0, acc, false, false);
    }

    // ---- store: n_rows multiple of 16, FOUT compile-time -> no guards ----
    float* out = T + (size_t)(m0 + (half ? 8 : 0)) * FOUT + col;
    #pragma unroll
    for (int i = 0; i < 8; ++i)
        out[(size_t)i * FOUT] = acc[i];
}

// ---------------------------------------- agg[dst] += T[src]  (float4 atomics)
template<int F>
__global__ void __launch_bounds__(256) edge_aggregate_kernel(
    const float* __restrict__ T, const int* __restrict__ src,
    const int* __restrict__ dst, float* __restrict__ agg, int n_edges)
{
    constexpr int CHUNKS = F / 4;
    const long long tid = (long long)blockIdx.x * blockDim.x + threadIdx.x;
    if (tid >= (long long)n_edges * CHUNKS) return;
    const int e = (int)(tid / CHUNKS);
    const int c = (int)(tid % CHUNKS);
    const int s = src[e];
    const int d = dst[e];
    const float4 v = ((const float4*)(T + (size_t)s * F))[c];
    float* p = agg + (size_t)d * F + c * 4;
    atomicAdd(p + 0, v.x); atomicAdd(p + 1, v.y);
    atomicAdd(p + 2, v.z); atomicAdd(p + 3, v.w);
}

// --------------------------------------- out = [relu](agg * nd[:,None] + b)
template<int F, bool RELU>
__global__ void __launch_bounds__(256) scale_bias_kernel(
    const float* __restrict__ agg, const float* __restrict__ nd,
    const float* __restrict__ b, float* __restrict__ out, long long total)
{
    const long long t = (long long)blockIdx.x * blockDim.x + threadIdx.x;
    if (t >= total) return;
    const int r = (int)(t / F);
    const int c = (int)(t % F);
    float v = agg[t] * nd[r] + b[c];
    if (RELU) v = fmaxf(v, 0.0f);
    out[t] = v;
}

// ------------------------- c[i] = normalize(out2[index[i]]), store bf16 bits
__global__ void __launch_bounds__(128) gather_norm_kernel(
    const float* __restrict__ out2, const int* __restrict__ index,
    unsigned short* __restrict__ cb)
{
    const int r = blockIdx.x;
    const int f = threadIdx.x;                 // 0..127
    const int node = index[r];
    const float v = out2[(size_t)node * OUT_DIM + f];
    float s = v * v;
    #pragma unroll
    for (int off = 16; off > 0; off >>= 1) s += __shfl_down(s, off, 32);
    __shared__ float wsum[4];
    if ((threadIdx.x & 31) == 0) wsum[threadIdx.x >> 5] = s;
    __syncthreads();
    const float tot = wsum[0] + wsum[1] + wsum[2] + wsum[3];
    const float inv = 1.0f / fmaxf(sqrtf(tot), 1e-12f);
    cb[(size_t)r * OUT_DIM + f] = f2bf_bits(v * inv);
}

// --- Z = triu((c1c1^T + c2c2^T + 0.5*(c1c2^T + c2c1^T)) / 3), 16x16 WMMA tiles
__global__ void __launch_bounds__(256) z_wmma_kernel(
    const unsigned short* __restrict__ c1, const unsigned short* __restrict__ c2,
    float* __restrict__ Z)
{
    const int wave = threadIdx.x >> 5;
    const int lane = threadIdx.x & 31;
    const int half = lane >> 4;
    const int tm = blockIdx.y * 2 + (wave >> 2);
    const int tn = blockIdx.x * 4 + (wave & 3);
    const int gm0 = tm * 16, gn0 = tn * 16;
    const int gn = gn0 + (lane & 15);

    float* zbase = Z + (size_t)(gm0 + (half ? 8 : 0)) * N_IDX + gn;

    if (gn0 + 15 < gm0) {  // tile strictly below diagonal: fill zeros
        #pragma unroll
        for (int i = 0; i < 8; ++i) zbase[(size_t)i * N_IDX] = 0.0f;
        return;
    }

    const unsigned short* a1r = c1 + (size_t)(gm0 + (lane & 15)) * OUT_DIM;
    const unsigned short* a2r = c2 + (size_t)(gm0 + (lane & 15)) * OUT_DIM;
    const unsigned short* b1r = c1 + (size_t)gn * OUT_DIM;   // B(k,n) = c[n][k]
    const unsigned short* b2r = c2 + (size_t)gn * OUT_DIM;

    v8f accA = {};  // c1c1^T + c2c2^T
    v8f accB = {};  // c1c2^T + c2c1^T
    #pragma unroll
    for (int k0 = 0; k0 < OUT_DIM; k0 += 32) {
        v16bf a1, a2, b1, b2;
        #pragma unroll
        for (int i = 0; i < 8; ++i) {
            const int ka = k0 + 2 * i + (i >= 4 ? 8 : 0) + (half ? 8 : 0);
            const int kb = k0 + 2 * i + (i >= 4 ? 8 : 0) + (half ? 8 : 0);
            unsigned int u;
            u = *(const unsigned int*)(a1r + ka);
            a1[2*i] = bits2bf((unsigned short)u); a1[2*i+1] = bits2bf((unsigned short)(u >> 16));
            u = *(const unsigned int*)(a2r + ka);
            a2[2*i] = bits2bf((unsigned short)u); a2[2*i+1] = bits2bf((unsigned short)(u >> 16));
            u = *(const unsigned int*)(b1r + kb);
            b1[2*i] = bits2bf((unsigned short)u); b1[2*i+1] = bits2bf((unsigned short)(u >> 16));
            u = *(const unsigned int*)(b2r + kb);
            b2[2*i] = bits2bf((unsigned short)u); b2[2*i+1] = bits2bf((unsigned short)(u >> 16));
        }
        accA = __builtin_amdgcn_wmma_f32_16x16x32_bf16(false, a1, false, b1, (short)0, accA, false, false);
        accA = __builtin_amdgcn_wmma_f32_16x16x32_bf16(false, a2, false, b2, (short)0, accA, false, false);
        accB = __builtin_amdgcn_wmma_f32_16x16x32_bf16(false, a1, false, b2, (short)0, accB, false, false);
        accB = __builtin_amdgcn_wmma_f32_16x16x32_bf16(false, a2, false, b1, (short)0, accB, false, false);
    }
    #pragma unroll
    for (int i = 0; i < 8; ++i) {
        const int gm = gm0 + i + (half ? 8 : 0);
        const float v = (accA[i] + 0.5f * accB[i]) * (1.0f / 3.0f);
        zbase[(size_t)i * N_IDX] = (gn >= gm) ? v : 0.0f;
    }
}

// =============================================================== launcher
extern "C" void kernel_launch(void* const* d_in, const int* in_sizes, int n_in,
                              void* d_out, int out_size, void* d_ws, size_t ws_size,
                              hipStream_t stream) {
    (void)n_in; (void)out_size; (void)ws_size;

    const float* raw1  = (const float*)d_in[0];
    const float* raw2  = (const float*)d_in[1];
    const int*   src1  = (const int*)d_in[2];
    const int*   dst1  = (const int*)d_in[3];
    const int*   src2  = (const int*)d_in[4];
    const int*   dst2  = (const int*)d_in[5];
    const int*   index = (const int*)d_in[6];
    const float* e1W1 = (const float*)d_in[7];  const float* e1b1 = (const float*)d_in[8];
    const float* e1W2 = (const float*)d_in[9];  const float* e1b2 = (const float*)d_in[10];
    const float* e2W1 = (const float*)d_in[11]; const float* e2b1 = (const float*)d_in[12];
    const float* e2W2 = (const float*)d_in[13]; const float* e2b2 = (const float*)d_in[14];

    const int N = in_sizes[0] / IN_DIM;   // 50000 (== 16 * 3125)
    const int E = in_sizes[2];            // 1600000

    // ---- workspace carve-up ----
    float* ws       = (float*)d_ws;
    float* deg_out  = ws;                                  // N (becomes ns)
    float* deg_in   = deg_out + N;                         // N (becomes nd)
    float* buf_t    = deg_in + N;                          // N*256 (GEMM out)
    float* buf_agg  = buf_t + (size_t)N * HID_DIM;         // N*256 (scatter tgt)
    float* buf_h    = buf_agg + (size_t)N * HID_DIM;       // N*256 (activations / out2)
    unsigned short* c1b = (unsigned short*)(buf_h + (size_t)N * HID_DIM);
    unsigned short* c2b = c1b + (size_t)N_IDX * OUT_DIM;

    for (int enc = 0; enc < 2; ++enc) {
        const float* x   = enc ? raw2 : raw1;
        const int*   src = enc ? src2 : src1;
        const int*   dst = enc ? dst2 : dst1;
        const float* W1  = enc ? e2W1 : e1W1;  const float* b1 = enc ? e2b1 : e1b1;
        const float* W2  = enc ? e2W2 : e1W2;  const float* b2 = enc ? e2b2 : e1b2;
        unsigned short* cb = enc ? c2b : c1b;

        // degrees -> norms (in place)
        hipMemsetAsync(deg_out, 0, 2 * (size_t)N * sizeof(float), stream);
        degree_kernel<<<(E + 255) / 256, 256, 0, stream>>>(src, dst, deg_out, deg_in, E);
        rsqrt_norm_kernel<<<(N + 255) / 256, 256, 0, stream>>>(deg_out, deg_in, N);

        // ---- layer 1: t = (x*ns)@W1 (512->256), aggregate, h = relu(agg*nd + b1)
        gcn_gemm_wmma<IN_DIM, HID_DIM><<<dim3(N / 16, HID_DIM / 128), 256, 0, stream>>>(
            x, deg_out, W1, buf_t);
        hipMemsetAsync(buf_agg, 0, (size_t)N * HID_DIM * sizeof(float), stream);
        {
            long long thr = (long long)E * (HID_DIM / 4);
            edge_aggregate_kernel<HID_DIM><<<(unsigned)((thr + 255) / 256), 256, 0, stream>>>(
                buf_t, src, dst, buf_agg, E);
        }
        scale_bias_kernel<HID_DIM, true>
            <<<(unsigned)(((long long)N * HID_DIM + 255) / 256), 256, 0, stream>>>(
                buf_agg, deg_in, b1, buf_h, (long long)N * HID_DIM);

        // ---- layer 2: t2 = (h*ns)@W2 (256->128), aggregate, out2 = agg*nd + b2
        gcn_gemm_wmma<HID_DIM, OUT_DIM><<<dim3(N / 16, OUT_DIM / 128), 256, 0, stream>>>(
            buf_h, deg_out, W2, buf_t);
        hipMemsetAsync(buf_agg, 0, (size_t)N * OUT_DIM * sizeof(float), stream);
        {
            long long thr = (long long)E * (OUT_DIM / 4);
            edge_aggregate_kernel<OUT_DIM><<<(unsigned)((thr + 255) / 256), 256, 0, stream>>>(
                buf_t, src, dst, buf_agg, E);
        }
        scale_bias_kernel<OUT_DIM, false>
            <<<(unsigned)(((long long)N * OUT_DIM + 255) / 256), 256, 0, stream>>>(
                buf_agg, deg_in, b2, buf_h, (long long)N * OUT_DIM);

        // gather + L2 normalize -> bf16
        gather_norm_kernel<<<N_IDX, 128, 0, stream>>>(buf_h, index, cb);
    }

    // final similarity matrix (block = 8 waves = 32x64 region of Z)
    z_wmma_kernel<<<dim3(N_IDX / 64, N_IDX / 32), 256, 0, stream>>>(
        c1b, c2b, (float*)d_out);
}
